// RPIANNmodel_7215545057504
// MI455X (gfx1250) — matrix operand order repaired and checked by reading, not compiled, not run on hardware
//
#include <hip/hip_runtime.h>
#include <cmath>

// ---------------------------------------------------------------------------
// RPIANN recursive predictor, bf16 WMMA implementation for gfx1250 (MI455X).
// All GEMMs run on v_wmma_f32_16x16x32_bf16 with f32 accumulation.
// 128x256 block tile, 8 waves of 64x64, double-buffered LDS fed by
// GLOBAL_LOAD_ASYNC_TO_LDS_B128 (ASYNCcnt) -- zero-VGPR staging.
// ---------------------------------------------------------------------------

typedef __attribute__((ext_vector_type(16))) __bf16 v16bf;
typedef __attribute__((ext_vector_type(8)))  __bf16 v8bf;
typedef __attribute__((ext_vector_type(8)))  float  v8f;

#define TILE_M 128
#define TILE_N 256
#define TILE_K 32

// ---- model dims (match reference) ----
#define BSZ   4096
#define D_IN  3072
#define DD    2048
#define HH    8192
#define D_OUT 1000
#define D_OUT_PAD 1024
#define NSTEPS 8
#define ACTION_SCALE 0.25f

struct Frag16 { v8bf lo, hi; };
__device__ __forceinline__ v16bf make_v16(v8bf lo, v8bf hi) {
    union { Frag16 p; v16bf v; } u;
    u.p.lo = lo; u.p.hi = hi;
    return u.v;
}

// Async DMA: global -> LDS, 16 bytes per lane, tracked by ASYNCcnt.
// VDST operand carries the LDS byte offset (low 32 bits of the generic
// pointer to __shared__ memory on AMDGPU), VADDR is the 64-bit global addr.
__device__ __forceinline__ void async_copy_b128(const __bf16* g, __bf16* l) {
    const unsigned loff = (unsigned)(unsigned long long)l;
    const unsigned long long ga = (unsigned long long)g;
    asm volatile("global_load_async_to_lds_b128 %0, %1, off"
                 :: "v"(loff), "v"(ga) : "memory");
}
__device__ __forceinline__ void wait_async_zero() {
    asm volatile("s_wait_asynccnt 0x0" ::: "memory");
}

// Epilogue variants
#define EPI_RELU_BF16      0   // outB = bf16(relu(acc))
#define EPI_BIAS_F32_BF16  1   // y = acc+bias (f32); outB = bf16(y)
#define EPI_BIAS_RELU_BF16 2   // outB = bf16(relu(acc+bias))
#define EPI_TANH_ACCUM     3   // y += tanh(acc+bias)*0.25; outB = bf16(y)
#define EPI_STORE_F32      4   // outF[col<Nreal] = acc   (logits)

template <int EPI>
__global__ __launch_bounds__(256)
void gemm_wmma_bf16(const __bf16* __restrict__ A, int lda,
                    const __bf16* __restrict__ W, int ldw,   // W is [N,K] row-major
                    const float*  __restrict__ bias,
                    float*        __restrict__ outF, int ldf,
                    __bf16*       __restrict__ outB, int ldb_,
                    float*        __restrict__ yPtr, int ldy,
                    int K, int Nreal)
{
    __shared__ __bf16 lA[2][TILE_M * TILE_K];   // 2 x 8 KB
    __shared__ __bf16 lB[2][TILE_N * TILE_K];   // 2 x 16 KB

    const int  tid  = threadIdx.x;
    const int  lane = tid & 31;
    const int  wave = tid >> 5;          // 8 waves
    const int  wm   = wave & 1;          // 2 along M (64 rows each)
    const int  wn   = wave >> 1;         // 4 along N (64 cols each)
    const long gm0  = (long)blockIdx.x * TILE_M;
    const long gn0  = (long)blockIdx.y * TILE_N;

    v8f acc[4][4];
#pragma unroll
    for (int i = 0; i < 4; ++i)
#pragma unroll
        for (int j = 0; j < 4; ++j)
#pragma unroll
            for (int e = 0; e < 8; ++e) acc[i][j][e] = 0.0f;

    // per-lane fragment coordinates (CDNA5 wave32 WMMA layouts)
    const int am  = lane & 15;           // row/col within 16
    const int kb  = (lane >> 4) * 8;     // unused; A reads also use kb2 (k = 16h + s)
    const int kb2 = (lane >> 4) * 16;    // B: K base {0,16}

    // DMA assignment: thread -> (row, 8-half segment) units
    int rowA[2], segA[2], rowB[4], segB[4];
#pragma unroll
    for (int u = 0; u < 2; ++u) {
        const int idx = tid + u * 256;              // 0..511 (128 rows x 4 segs)
        rowA[u] = idx >> 2; segA[u] = (idx & 3) * 8;
    }
#pragma unroll
    for (int u = 0; u < 4; ++u) {
        const int idx = tid + u * 256;              // 0..1023 (256 rows x 4 segs)
        rowB[u] = idx >> 2; segB[u] = (idx & 3) * 8;
    }

    auto fetch_async = [&](int k0, int buf) {
#pragma unroll
        for (int u = 0; u < 2; ++u)
            async_copy_b128(A + (gm0 + rowA[u]) * (long)lda + k0 + segA[u],
                            &lA[buf][rowA[u] * TILE_K + segA[u]]);
#pragma unroll
        for (int u = 0; u < 4; ++u)
            async_copy_b128(W + (gn0 + rowB[u]) * (long)ldw + k0 + segB[u],
                            &lB[buf][rowB[u] * TILE_K + segB[u]]);
    };
    auto compute = [&](int buf) {
        v16bf bfr[4];
#pragma unroll
        for (int nt = 0; nt < 4; ++nt) {
            const int n = wn * 64 + nt * 16 + am;
            bfr[nt] = *(const v16bf*)(&lB[buf][n * TILE_K + kb2]);
        }
#pragma unroll
        for (int mt = 0; mt < 4; ++mt) {
            const int m = wm * 64 + mt * 16 + am;
            const v8bf alo = *(const v8bf*)(&lA[buf][m * TILE_K + kb2]);
            const v8bf ahi = *(const v8bf*)(&lA[buf][m * TILE_K + kb2 + 8]);
            const v16bf afr = make_v16(alo, ahi);
#pragma unroll
            for (int nt = 0; nt < 4; ++nt) {
                acc[mt][nt] = __builtin_amdgcn_wmma_f32_16x16x32_bf16(
                    false, afr, false, bfr[nt], (short)0, acc[mt][nt],
                    false, false);
            }
        }
    };

    // ---- pipeline prologue: DMA chunk 0 into buffer 0 ----
    fetch_async(0, 0);
    wait_async_zero();
    __syncthreads();

    int cur = 0;
    for (int k0 = TILE_K; k0 < K; k0 += TILE_K) {
        fetch_async(k0, cur ^ 1);   // DMA next chunk (ASYNCcnt in flight)...
        compute(cur);               // ...overlapped with 16 WMMAs
        wait_async_zero();          // this wave's DMA landed in LDS
        __syncthreads();            // all waves' DMA landed
        cur ^= 1;
    }
    compute(cur);

    // ---- epilogue: C/D layout -> (row,col) ----
    const int colL = lane & 15;
    const int rowS = (lane >> 4) * 8;
#pragma unroll
    for (int nt = 0; nt < 4; ++nt) {
        const long col = gn0 + wn * 64 + nt * 16 + colL;
        float bv = 0.0f;
        if constexpr (EPI == EPI_BIAS_F32_BF16 || EPI == EPI_BIAS_RELU_BF16 ||
                      EPI == EPI_TANH_ACCUM) {
            bv = bias[col];
        }
#pragma unroll
        for (int mt = 0; mt < 4; ++mt) {
#pragma unroll
            for (int v = 0; v < 8; ++v) {
                const long row = gm0 + wm * 64 + mt * 16 + rowS + v;
                const float a = acc[mt][nt][v] + bv;
                if constexpr (EPI == EPI_RELU_BF16) {
                    const float r = a > 0.0f ? a : 0.0f;
                    outB[row * (long)ldb_ + col] = (__bf16)r;
                } else if constexpr (EPI == EPI_BIAS_F32_BF16) {
                    yPtr[row * (long)ldy + col] = a;
                    outB[row * (long)ldb_ + col] = (__bf16)a;
                } else if constexpr (EPI == EPI_BIAS_RELU_BF16) {
                    const float r = a > 0.0f ? a : 0.0f;
                    outB[row * (long)ldb_ + col] = (__bf16)r;
                } else if constexpr (EPI == EPI_TANH_ACCUM) {
                    const float up = tanhf(a) * ACTION_SCALE;
                    const float yn = yPtr[row * (long)ldy + col] + up;
                    yPtr[row * (long)ldy + col] = yn;
                    outB[row * (long)ldb_ + col] = (__bf16)yn;
                } else { // EPI_STORE_F32
                    if (col < Nreal) outF[row * (long)ldf + col] = a;
                }
            }
        }
    }
}

// ---------------------------------------------------------------------------
// Elementwise helpers
// ---------------------------------------------------------------------------
__global__ __launch_bounds__(256)
void cvt_f32_bf16(const float* __restrict__ s, __bf16* __restrict__ d, long n)
{
    const long stride = (long)gridDim.x * blockDim.x;
    for (long i = (long)blockIdx.x * blockDim.x + threadIdx.x; i < n; i += stride)
        d[i] = (__bf16)s[i];
}

__global__ __launch_bounds__(256)
void relu_f32_bf16(const float* __restrict__ s, __bf16* __restrict__ d, long n)
{
    const long stride = (long)gridDim.x * blockDim.x;
    for (long i = (long)blockIdx.x * blockDim.x + threadIdx.x; i < n; i += stride) {
        const float v = s[i];
        d[i] = (__bf16)(v > 0.0f ? v : 0.0f);
    }
}

// W_tgt [D, D_OUT] f32 -> Wt [D_OUT_PAD, D] bf16 (transposed, zero-padded rows)
__global__ __launch_bounds__(256)
void transpose_pad_wtgt(const float* __restrict__ s, __bf16* __restrict__ d)
{
    const long n = (long)D_OUT_PAD * DD;
    const long stride = (long)gridDim.x * blockDim.x;
    for (long i = (long)blockIdx.x * blockDim.x + threadIdx.x; i < n; i += stride) {
        const long nrow = i / DD;       // padded output index
        const long k    = i - nrow * DD;
        d[i] = (nrow < D_OUT) ? (__bf16)s[k * D_OUT + nrow] : (__bf16)0.0f;
    }
}

// ---------------------------------------------------------------------------
// Host-side orchestration
// ---------------------------------------------------------------------------
extern "C" void kernel_launch(void* const* d_in, const int* in_sizes, int n_in,
                              void* d_out, int out_size, void* d_ws, size_t ws_size,
                              hipStream_t stream)
{
    (void)in_sizes; (void)n_in; (void)out_size; (void)ws_size;

    const float* x      = (const float*)d_in[0];  // [B, D_IN]
    const float* W_in   = (const float*)d_in[1];  // [D, D_IN]
    const float* W_tgt  = (const float*)d_in[2];  // [D, D_OUT]
    const float* W_pred = (const float*)d_in[3];  // [D, D]
    const float* b_pred = (const float*)d_in[4];  // [D]
    const float* W1     = (const float*)d_in[5];  // [H, 2D]
    const float* b1     = (const float*)d_in[6];  // [H]
    const float* W2     = (const float*)d_in[7];  // [D, H]
    const float* b2     = (const float*)d_in[8];  // [D]
    float* out = (float*)d_out;                   // [B, D_OUT]

    // workspace layout (bf16 unless noted)
    char* ws = (char*)d_ws;
    size_t off = 0;
    auto alloc = [&](size_t bytes) { char* p = ws + off; off += (bytes + 255) & ~(size_t)255; return p; };
    __bf16* xb       = (__bf16*)alloc((size_t)BSZ * D_IN * 2);       // x bf16
    __bf16* Winb     = (__bf16*)alloc((size_t)DD * D_IN * 2);
    __bf16* Wpredb   = (__bf16*)alloc((size_t)DD * DD * 2);
    __bf16* W1b      = (__bf16*)alloc((size_t)HH * 2 * DD * 2);
    __bf16* W2b      = (__bf16*)alloc((size_t)DD * HH * 2);
    __bf16* Wtgtb    = (__bf16*)alloc((size_t)D_OUT_PAD * DD * 2);   // transposed+padded
    __bf16* combined = (__bf16*)alloc((size_t)BSZ * 2 * DD * 2);     // [B, 2D] bf16
    float*  y        = (float*) alloc((size_t)BSZ * DD * 4);         // y_hat f32
    __bf16* hb       = (__bf16*)alloc((size_t)BSZ * HH * 2);         // hidden bf16
    __bf16* yrelu    = (__bf16*)alloc((size_t)BSZ * DD * 2);         // relu(y) bf16

    const dim3 blk(256);
    const dim3 cvtGrid(2048);

    // 1) precision conversion of inputs/weights
    cvt_f32_bf16<<<cvtGrid, blk, 0, stream>>>(x,      xb,     (long)BSZ * D_IN);
    cvt_f32_bf16<<<cvtGrid, blk, 0, stream>>>(W_in,   Winb,   (long)DD * D_IN);
    cvt_f32_bf16<<<cvtGrid, blk, 0, stream>>>(W_pred, Wpredb, (long)DD * DD);
    cvt_f32_bf16<<<cvtGrid, blk, 0, stream>>>(W1,     W1b,    (long)HH * 2 * DD);
    cvt_f32_bf16<<<cvtGrid, blk, 0, stream>>>(W2,     W2b,    (long)DD * HH);
    transpose_pad_wtgt<<<cvtGrid, blk, 0, stream>>>(W_tgt, Wtgtb);

    // 2) x_embed = relu(x @ W_in^T)  -> combined[:, 0:D]
    {
        dim3 grid(BSZ / TILE_M, DD / TILE_N);
        gemm_wmma_bf16<EPI_RELU_BF16><<<grid, blk, 0, stream>>>(
            xb, D_IN, Winb, D_IN, nullptr,
            nullptr, 0, combined, 2 * DD, nullptr, 0, D_IN, DD);
    }
    // 3) y0 = x_embed @ W_pred^T + b_pred   (f32 y, bf16 into combined[:, D:2D])
    {
        dim3 grid(BSZ / TILE_M, DD / TILE_N);
        gemm_wmma_bf16<EPI_BIAS_F32_BF16><<<grid, blk, 0, stream>>>(
            combined, 2 * DD, Wpredb, DD, b_pred,
            nullptr, 0, combined + DD, 2 * DD, y, DD, DD, DD);
    }
    // 4) 8 recursive steps
    for (int s = 0; s < NSTEPS; ++s) {
        {   // h = relu(combined @ W1^T + b1)
            dim3 grid(BSZ / TILE_M, HH / TILE_N);
            gemm_wmma_bf16<EPI_BIAS_RELU_BF16><<<grid, blk, 0, stream>>>(
                combined, 2 * DD, W1b, 2 * DD, b1,
                nullptr, 0, hb, HH, nullptr, 0, 2 * DD, HH);
        }
        {   // y += tanh(h @ W2^T + b2) * 0.25 ; combined[:, D:2D] = bf16(y)
            dim3 grid(BSZ / TILE_M, DD / TILE_N);
            gemm_wmma_bf16<EPI_TANH_ACCUM><<<grid, blk, 0, stream>>>(
                hb, HH, W2b, HH, b2,
                nullptr, 0, combined + DD, 2 * DD, y, DD, HH, DD);
        }
    }
    // 5) logits = relu(y) @ W_tgt  (W pre-transposed/padded to [1024, D])
    relu_f32_bf16<<<cvtGrid, blk, 0, stream>>>(y, yrelu, (long)BSZ * DD);
    {
        dim3 grid(BSZ / TILE_M, D_OUT_PAD / TILE_N);
        gemm_wmma_bf16<EPI_STORE_F32><<<grid, blk, 0, stream>>>(
            yrelu, DD, Wtgtb, DD, nullptr,
            out, D_OUT, nullptr, 0, nullptr, 0, DD, D_OUT);
    }
}
